// TextSwitchTransformer_10831907521072
// MI455X (gfx1250) — compile-verified, matrix operand order checked by reading, not run on hardware
//
#include <hip/hip_runtime.h>
#include <hip/hip_bf16.h>
#include <math.h>

// ---------------- model constants ----------------
#define VOC   32000
#define Dm    1024
#define Hh    16
#define HD    64
#define Ll    6
#define Ee    8
#define DFFc  2048
#define Bb    8
#define Ss    512
#define Tt    (Bb*Ss)        // 4096 tokens
#define CAPc  614            // int(1.2*4096/8)

typedef __attribute__((ext_vector_type(16))) __bf16 v16bf;
typedef __attribute__((ext_vector_type(8)))  __bf16 v8bf;
typedef __attribute__((ext_vector_type(8)))  float  v8f;

// ---------------- generic batched WMMA GEMM ----------------
// Requirements (hold at every call site): N % 64 == 0, K % 32 == 0.
// Only M may be ragged; out-of-range A rows are clamped (outputs discarded
// by the epilogue guard), so all tile loads are branch-free.
// B layout: transB ? B[n*ldb+k] : B[k*ldb+n]
// per-z offsets: ptr += (z/zdiv)*bs_hi + (z%zdiv)*bs_lo   (bias += z*bsBias)
#define BMt 64
#define BNt 64
#define BKt 32

__global__ __launch_bounds__(256) void gemm_wmma(
    const float* __restrict__ A, const float* __restrict__ Bm, float* __restrict__ C,
    const float* __restrict__ bias, const float* __restrict__ residual,
    int M, int N, int K, int lda, int ldb, int ldc,
    int transB, int relu, float alpha,
    long long bsA_hi, long long bsA_lo,
    long long bsB_hi, long long bsB_lo,
    long long bsC_hi, long long bsC_lo,
    long long bsBias, int zdiv)
{
  const int zb = blockIdx.z;
  const long long zh = zb / zdiv, zl = zb % zdiv;
  A  += zh * bsA_hi + zl * bsA_lo;
  Bm += zh * bsB_hi + zl * bsB_lo;
  C  += zh * bsC_hi + zl * bsC_lo;
  if (bias) bias += (long long)zb * bsBias;

  // LDS tiles stored directly in WMMA fragment order: [tile16][lane][16 halves]
  __shared__ __bf16 As2[4][32][16];
  __shared__ __bf16 Bs2[4][32][16];

  const int tid  = threadIdx.x;
  const int wid  = tid >> 5;
  const int lane = tid & 31;
  const int lrow = lane & 15;
  const int g    = lane >> 4;
  const int wm   = wid & 3;   // m-tile 0..3 (4*16 = 64 rows)
  const int wn   = wid >> 2;  // n-half 0..1 (2*32 = 64 cols)

  const int row0 = blockIdx.x * BMt;
  const int col0 = blockIdx.y * BNt;

  v8f acc[2];
#pragma unroll
  for (int r = 0; r < 8; ++r) { acc[0][r] = 0.f; acc[1][r] = 0.f; }

  // ---- A loader mapping: 64 rows x (4 groups of 8 k) ----
  const int a_row  = tid >> 2;               // 0..63
  const int a_j    = tid & 3;                // k-group
  const int a_mt   = a_row >> 4;
  const int a_lane = (a_row & 15) + ((a_j & 1) << 4);
  const int a_eoff = (a_j >> 1) << 3;        // 0 or 8
  const int a_m    = min(row0 + a_row, M - 1);        // clamp ragged M
  const float* Aptr = A + (size_t)a_m * lda + a_j * 8;

  // ---- B loader mappings ----
  // transB: identical shape to A loader over n-rows
  const int bt_n    = tid >> 2;              // 0..63 (n within block)
  const int bt_j    = tid & 3;
  const int bt_nt   = bt_n >> 4;
  const int bt_lane = (bt_n & 15) + ((bt_j & 1) << 4);
  const int bt_eoff = (bt_j >> 1) << 3;
  const float* Btp = transB ? (Bm + (size_t)(col0 + bt_n) * ldb + bt_j * 8) : nullptr;
  // non-trans: 32 k-rows x (8 groups of 8 n)
  const int bn_kk  = tid >> 3;               // 0..31
  const int bn_nn  = (tid & 7) * 8;          // 0..56
  const int bn_g   = (bn_kk >> 3) & 1;
  const int bn_e   = ((bn_kk >> 4) << 3) + (bn_kk & 7);
  const float* Bnp = transB ? nullptr : (Bm + (size_t)bn_kk * ldb + col0 + bn_nn);

  for (int k0 = 0; k0 < K; k0 += BKt) {
    // ---- stage A (two float4 -> one 16B fragment-order LDS store) ----
    {
      const float4 f0 = *reinterpret_cast<const float4*>(Aptr + k0);
      const float4 f1 = *reinterpret_cast<const float4*>(Aptr + k0 + 4);
      v8bf pk;
      pk[0] = (__bf16)(f0.x * alpha); pk[1] = (__bf16)(f0.y * alpha);
      pk[2] = (__bf16)(f0.z * alpha); pk[3] = (__bf16)(f0.w * alpha);
      pk[4] = (__bf16)(f1.x * alpha); pk[5] = (__bf16)(f1.y * alpha);
      pk[6] = (__bf16)(f1.z * alpha); pk[7] = (__bf16)(f1.w * alpha);
      *reinterpret_cast<v8bf*>(&As2[a_mt][a_lane][a_eoff]) = pk;
    }
    // ---- stage B ----
    if (transB) {
      const float4 f0 = *reinterpret_cast<const float4*>(Btp + k0);
      const float4 f1 = *reinterpret_cast<const float4*>(Btp + k0 + 4);
      v8bf pk;
      pk[0] = (__bf16)f0.x; pk[1] = (__bf16)f0.y; pk[2] = (__bf16)f0.z; pk[3] = (__bf16)f0.w;
      pk[4] = (__bf16)f1.x; pk[5] = (__bf16)f1.y; pk[6] = (__bf16)f1.z; pk[7] = (__bf16)f1.w;
      *reinterpret_cast<v8bf*>(&Bs2[bt_nt][bt_lane][bt_eoff]) = pk;
    } else {
      const float4 f0 = *reinterpret_cast<const float4*>(Bnp + (size_t)k0 * ldb);
      const float4 f1 = *reinterpret_cast<const float4*>(Bnp + (size_t)k0 * ldb + 4);
      float vv[8] = {f0.x, f0.y, f0.z, f0.w, f1.x, f1.y, f1.z, f1.w};
#pragma unroll
      for (int i = 0; i < 8; ++i) {
        const int n = bn_nn + i;
        Bs2[n >> 4][(n & 15) + (bn_g << 4)][bn_e] = (__bf16)vv[i];
      }
    }
    // prefetch next K-tile of the streaming operands (-> global_prefetch_b8)
    if (k0 + BKt < K) {
      __builtin_prefetch(Aptr + k0 + BKt, 0, 1);
      if (transB) __builtin_prefetch(Btp + k0 + BKt, 0, 1);
      else        __builtin_prefetch(Bnp + (size_t)(k0 + BKt) * ldb, 0, 1);
    }
    __syncthreads();

    // ---- fragments: one 32B vector LDS load each ----
    const v16bf af  = *reinterpret_cast<const v16bf*>(&As2[wm][lane][0]);
    const v16bf bf0 = *reinterpret_cast<const v16bf*>(&Bs2[wn * 2 + 0][lane][0]);
    const v16bf bf1 = *reinterpret_cast<const v16bf*>(&Bs2[wn * 2 + 1][lane][0]);
    acc[0] = __builtin_amdgcn_wmma_f32_16x16x32_bf16(false, af, false, bf0,
                                                     (short)0, acc[0], false, false);
    acc[1] = __builtin_amdgcn_wmma_f32_16x16x32_bf16(false, af, false, bf1,
                                                     (short)0, acc[1], false, false);
    __syncthreads();
  }

  // ---- epilogue: C-layout lane(l): N = l%16, VGPR r: M = r + 8*(l/16) ----
#pragma unroll
  for (int f = 0; f < 2; ++f) {
#pragma unroll
    for (int r = 0; r < 8; ++r) {
      const int mg = row0 + wm * 16 + r + 8 * g;
      const int ng = col0 + wn * 32 + f * 16 + lrow;
      if (mg < M) {
        float v = acc[f][r];
        if (bias)     v += bias[ng];
        if (residual) v += residual[(size_t)mg * ldc + ng];
        if (relu)     v = fmaxf(v, 0.f);
        C[(size_t)mg * ldc + ng] = v;
      }
    }
  }
}

// ---------------- embedding + sinusoidal PE ----------------
__global__ void embed_pe(const int* __restrict__ ids, const float* __restrict__ emb,
                         float* __restrict__ x)
{
  const int t = blockIdx.x;                  // token row t = s*B + b
  const int d = blockIdx.y * 256 + threadIdx.x;
  const int s = t / Bb, b = t % Bb;
  const int id = ids[b * Ss + s];
  const int i = d >> 1;
  const float div = __expf(-(float)(2 * i) * (9.210340371976184f / (float)Dm));
  const float ang = (float)s * div;
  const float pe = (d & 1) ? __cosf(ang) : __sinf(ang);
  x[(size_t)t * Dm + d] = emb[(size_t)id * Dm + d] + pe;
}

// ---------------- LayerNorm (optional [S,B]->[B,S] transposed write) ----
__global__ void layernorm_k(const float* __restrict__ x, const float* __restrict__ sc,
                            const float* __restrict__ bi, float* __restrict__ out,
                            int transpose)
{
  const int t = blockIdx.x;
  const int tid = threadIdx.x;
  __shared__ float red[256];
  const float* row = x + (size_t)t * Dm;

  float s = 0.f;
  for (int d = tid; d < Dm; d += 256) s += row[d];
  red[tid] = s; __syncthreads();
  for (int o = 128; o > 0; o >>= 1) { if (tid < o) red[tid] += red[tid + o]; __syncthreads(); }
  const float mu = red[0] * (1.0f / Dm);
  __syncthreads();

  float vs = 0.f;
  for (int d = tid; d < Dm; d += 256) { float df = row[d] - mu; vs += df * df; }
  red[tid] = vs; __syncthreads();
  for (int o = 128; o > 0; o >>= 1) { if (tid < o) red[tid] += red[tid + o]; __syncthreads(); }
  const float rstd = rsqrtf(red[0] * (1.0f / Dm) + 1e-5f);

  const size_t orow = transpose ? ((size_t)(t % Bb) * Ss + (size_t)(t / Bb)) : (size_t)t;
  for (int d = tid; d < Dm; d += 256)
    out[orow * Dm + d] = (row[d] - mu) * rstd * sc[d] + bi[d];
}

// ---------------- row softmax (attention, in place) ----------------
__global__ void softmax_rows(float* __restrict__ a, int R)
{
  float* p = a + (size_t)blockIdx.x * R;
  const int tid = threadIdx.x;
  __shared__ float red[256];

  float m = -3.0e38f;
  for (int i = tid; i < R; i += 256) m = fmaxf(m, p[i]);
  red[tid] = m; __syncthreads();
  for (int o = 128; o > 0; o >>= 1) { if (tid < o) red[tid] = fmaxf(red[tid], red[tid + o]); __syncthreads(); }
  m = red[0]; __syncthreads();

  float s = 0.f;
  for (int i = tid; i < R; i += 256) { float e = __expf(p[i] - m); p[i] = e; s += e; }
  red[tid] = s; __syncthreads();
  for (int o = 128; o > 0; o >>= 1) { if (tid < o) red[tid] += red[tid + o]; __syncthreads(); }
  const float inv = 1.0f / red[0];
  for (int i = tid; i < R; i += 256) p[i] *= inv;
}

// ---------------- router: softmax(z @ rw.T + rb), argmax, pmax ------
__global__ void router_k(const float* __restrict__ z, const float* __restrict__ rw,
                         const float* __restrict__ rb, float* __restrict__ probs,
                         float* __restrict__ pmax, int* __restrict__ route)
{
  const int t = blockIdx.x * 256 + threadIdx.x;
  if (t >= Tt) return;
  const float* r = z + (size_t)t * Dm;
  float lg[Ee];
#pragma unroll
  for (int e = 0; e < Ee; ++e) {
    const float* w = rw + (size_t)e * Dm;
    float acc = rb[e];
    for (int d = 0; d < Dm; ++d) acc += r[d] * w[d];
    lg[e] = acc;
  }
  float m = lg[0]; int am = 0;
#pragma unroll
  for (int e = 1; e < Ee; ++e) if (lg[e] > m) { m = lg[e]; am = e; }
  float s = 0.f;
#pragma unroll
  for (int e = 0; e < Ee; ++e) { lg[e] = __expf(lg[e] - m); s += lg[e]; }
  const float inv = 1.0f / s;
#pragma unroll
  for (int e = 0; e < Ee; ++e) probs[(size_t)t * Ee + e] = lg[e] * inv;
  pmax[t]  = lg[am] * inv;
  route[t] = am;
}

// ------- deterministic capacity scan (matches JAX cumsum drop order) ----
__global__ void route_scan(const float* __restrict__ probs, const float* __restrict__ pmax,
                           const int* __restrict__ route, int* __restrict__ slot,
                           int* __restrict__ keep, float* __restrict__ out_counts,
                           float* __restrict__ out_rp, float* __restrict__ out_nd,
                           float* __restrict__ out_pm)
{
  const int tid = threadIdx.x;
  for (int t = tid; t < Tt; t += 256) out_pm[t] = pmax[t];
  if (tid < Ee) {
    float s = 0.f;
    for (int t = 0; t < Tt; ++t) s += probs[(size_t)t * Ee + tid];
    out_rp[tid] = s;
  }
  if (tid == 0) {
    int run[Ee]; for (int e = 0; e < Ee; ++e) run[e] = 0;
    int nd = 0;
    for (int t = 0; t < Tt; ++t) {
      const int e = route[t];
      const int pos = run[e]++;
      const int k = (pos < CAPc) ? 1 : 0;
      keep[t] = k;
      slot[t] = k ? (e * CAPc + pos) : 0;
      if (!k) nd++;
    }
    for (int e = 0; e < Ee; ++e) out_counts[e] = (float)run[e];
    out_nd[0] = (float)nd;
  }
}

// ---------------- scatter kept tokens into expert buffer ----------------
__global__ void scatter_k(const float* __restrict__ z, const int* __restrict__ slot,
                          const int* __restrict__ keep, float* __restrict__ buf)
{
  const int t = blockIdx.x;
  const int d = blockIdx.y * 256 + threadIdx.x;
  if (keep[t]) buf[(size_t)slot[t] * Dm + d] = z[(size_t)t * Dm + d];
}

// -------- gather expert output (or passthrough), scale by pmax, residual ----
__global__ void gather_k(float* __restrict__ x, const float* __restrict__ z,
                         const float* __restrict__ eo, const int* __restrict__ slot,
                         const int* __restrict__ keep, const float* __restrict__ pmax)
{
  const int t = blockIdx.x;
  const int d = blockIdx.y * 256 + threadIdx.x;
  const float v = keep[t] ? eo[(size_t)slot[t] * Dm + d] : z[(size_t)t * Dm + d];
  x[(size_t)t * Dm + d] += v * pmax[t];
}

// ---------------- in-place log-softmax over vocab ----------------
__global__ void logsoftmax_k(float* __restrict__ a)
{
  float* p = a + (size_t)blockIdx.x * VOC;
  const int tid = threadIdx.x;
  __shared__ float red[256];

  float m = -3.0e38f;
  for (int i = tid; i < VOC; i += 256) m = fmaxf(m, p[i]);
  red[tid] = m; __syncthreads();
  for (int o = 128; o > 0; o >>= 1) { if (tid < o) red[tid] = fmaxf(red[tid], red[tid + o]); __syncthreads(); }
  m = red[0]; __syncthreads();

  float s = 0.f;
  for (int i = tid; i < VOC; i += 256) s += __expf(p[i] - m);
  red[tid] = s; __syncthreads();
  for (int o = 128; o > 0; o >>= 1) { if (tid < o) red[tid] += red[tid + o]; __syncthreads(); }
  const float lse = m + __logf(red[0]);
  for (int i = tid; i < VOC; i += 256) p[i] = p[i] - lse;
}

// ============================== host ==============================
extern "C" void kernel_launch(void* const* d_in, const int* in_sizes, int n_in,
                              void* d_out, int out_size, void* d_ws, size_t ws_size,
                              hipStream_t stream)
{
  const int*   ids   = (const int*)  d_in[0];
  const float* embed = (const float*)d_in[1];
  const float* ln1_s = (const float*)d_in[2];
  const float* ln1_b = (const float*)d_in[3];
  const float* in_w  = (const float*)d_in[4];
  const float* in_b  = (const float*)d_in[5];
  const float* out_w = (const float*)d_in[6];
  const float* out_b = (const float*)d_in[7];
  const float* ln2_s = (const float*)d_in[8];
  const float* ln2_b = (const float*)d_in[9];
  const float* rw    = (const float*)d_in[10];
  const float* rb    = (const float*)d_in[11];
  const float* w1    = (const float*)d_in[12];
  const float* b1    = (const float*)d_in[13];
  const float* w2    = (const float*)d_in[14];
  const float* b2    = (const float*)d_in[15];
  const float* lnf_s = (const float*)d_in[16];
  const float* lnf_b = (const float*)d_in[17];
  const float* dec_w = (const float*)d_in[18];
  const float* dec_b = (const float*)d_in[19];

  // ---- workspace carve-out ----
  float* p = (float*)d_ws;
  float* x     = p; p += (size_t)Tt * Dm;
  float* z     = p; p += (size_t)Tt * Dm;
  float* qkv   = p; p += (size_t)Tt * 3 * Dm;
  float* att   = p; p += (size_t)Bb * Hh * Ss * Ss;
  float* ao    = p; p += (size_t)Tt * Dm;
  float* probs = p; p += (size_t)Tt * Ee;
  float* pmax  = p; p += (size_t)Tt;
  int*   route = (int*)p; p += Tt;
  int*   slot  = (int*)p; p += Tt;
  int*   keep  = (int*)p; p += Tt;
  float* buf   = p; p += (size_t)Ee * CAPc * Dm;
  float* hb    = p; p += (size_t)Ee * CAPc * DFFc;
  float* eo    = p; p += (size_t)Ee * CAPc * Dm;
  float* yT    = p; p += (size_t)Tt * Dm;

  // ---- output carve-out (flat, return order) ----
  float* out_lp     = (float*)d_out;                       // [B,S,VOC]
  float* out_counts = out_lp + (size_t)Bb * Ss * VOC;      // [L,E]
  float* out_rp     = out_counts + (size_t)Ll * Ee;        // [L,E]
  float* out_nd     = out_rp + (size_t)Ll * Ee;            // [L]
  float* out_pm     = out_nd + Ll;                         // [L,T]

  const dim3 blk(256);

  // x = embed[ids] + PE   (rows t = s*B + b)
  embed_pe<<<dim3(Tt, Dm / 256), blk, 0, stream>>>(ids, embed, x);

  for (int l = 0; l < Ll; ++l) {
    // ---- LN1 ----
    layernorm_k<<<Tt, blk, 0, stream>>>(x, ln1_s + (size_t)l * Dm, ln1_b + (size_t)l * Dm, z, 0);

    // ---- QKV = z @ in_w[l].T + in_b[l] ----
    gemm_wmma<<<dim3(Tt / BMt, 3 * Dm / BNt, 1), blk, 0, stream>>>(
        z, in_w + (size_t)l * 3 * Dm * Dm, qkv,
        in_b + (size_t)l * 3 * Dm, nullptr,
        Tt, 3 * Dm, Dm, Dm, Dm, 3 * Dm,
        1, 0, 1.0f, 0, 0, 0, 0, 0, 0, 0, 1);

    // ---- scores = (q*scale) @ k^T per (b,h); z-batch = b*H+h ----
    gemm_wmma<<<dim3(Ss / BMt, Ss / BNt, Bb * Hh), blk, 0, stream>>>(
        qkv, qkv + Dm, att, nullptr, nullptr,
        Ss, Ss, HD, Bb * 3 * Dm, Bb * 3 * Dm, Ss,
        1, 0, 0.125f,
        3 * Dm, HD,
        3 * Dm, HD,
        (long long)Hh * Ss * Ss, (long long)Ss * Ss,
        0, Hh);

    // ---- softmax over keys ----
    softmax_rows<<<Bb * Hh * Ss, blk, 0, stream>>>(att, Ss);

    // ---- out = att @ v ; written into merged-head layout [t, h*64+d] ----
    gemm_wmma<<<dim3(Ss / BMt, 1, Bb * Hh), blk, 0, stream>>>(
        att, qkv + 2 * Dm, ao, nullptr, nullptr,
        Ss, HD, Ss, Ss, Bb * 3 * Dm, Bb * Dm,
        0, 0, 1.0f,
        (long long)Hh * Ss * Ss, (long long)Ss * Ss,
        3 * Dm, HD,
        Dm, HD,
        0, Hh);

    // ---- x = x + ao @ out_w[l].T + out_b[l] ----
    gemm_wmma<<<dim3(Tt / BMt, Dm / BNt, 1), blk, 0, stream>>>(
        ao, out_w + (size_t)l * Dm * Dm, x,
        out_b + (size_t)l * Dm, x,
        Tt, Dm, Dm, Dm, Dm, Dm,
        1, 0, 1.0f, 0, 0, 0, 0, 0, 0, 0, 1);

    // ---- LN2 ----
    layernorm_k<<<Tt, blk, 0, stream>>>(x, ln2_s + (size_t)l * Dm, ln2_b + (size_t)l * Dm, z, 0);

    // ---- router + deterministic capacity routing ----
    router_k<<<Tt / 256, blk, 0, stream>>>(z, rw + (size_t)l * Ee * Dm, rb + (size_t)l * Ee,
                                           probs, pmax, route);
    hipMemsetAsync(buf, 0, (size_t)Ee * CAPc * Dm * sizeof(float), stream);
    route_scan<<<1, blk, 0, stream>>>(probs, pmax, route, slot, keep,
                                      out_counts + (size_t)l * Ee,
                                      out_rp + (size_t)l * Ee,
                                      out_nd + l,
                                      out_pm + (size_t)l * Tt);
    scatter_k<<<dim3(Tt, Dm / 256), blk, 0, stream>>>(z, slot, keep, buf);

    // ---- expert FFN1: relu(buf @ w1[l,e] + b1[l,e]) ; z-batch = expert ----
    gemm_wmma<<<dim3((CAPc + BMt - 1) / BMt, DFFc / BNt, Ee), blk, 0, stream>>>(
        buf, w1 + (size_t)l * Ee * Dm * DFFc, hb,
        b1 + (size_t)l * Ee * DFFc, nullptr,
        CAPc, DFFc, Dm, Dm, DFFc, DFFc,
        0, 1, 1.0f,
        (long long)CAPc * Dm, 0,
        (long long)Dm * DFFc, 0,
        (long long)CAPc * DFFc, 0,
        DFFc, 1);

    // ---- expert FFN2: h @ w2[l,e] + b2[l,e] ----
    gemm_wmma<<<dim3((CAPc + BMt - 1) / BMt, Dm / BNt, Ee), blk, 0, stream>>>(
        hb, w2 + (size_t)l * Ee * DFFc * Dm, eo,
        b2 + (size_t)l * Ee * Dm, nullptr,
        CAPc, Dm, DFFc, DFFc, Dm, Dm,
        0, 0, 1.0f,
        (long long)CAPc * DFFc, 0,
        (long long)DFFc * Dm, 0,
        (long long)CAPc * Dm, 0,
        Dm, 1);

    // ---- x += pmax * (kept ? expert_out : z) ----
    gather_k<<<dim3(Tt, Dm / 256), blk, 0, stream>>>(x, z, eo, slot, keep, pmax);
  }

  // ---- final LN, written transposed to [B,S,D] token order ----
  layernorm_k<<<Tt, blk, 0, stream>>>(x, lnf_s, lnf_b, yT, 1);

  // ---- logits = yT @ dec_w.T + dec_b  -> directly into d_out ----
  gemm_wmma<<<dim3(Tt / BMt, VOC / BNt, 1), blk, 0, stream>>>(
      yT, dec_w, out_lp, dec_b, nullptr,
      Tt, VOC, Dm, Dm, Dm, VOC,
      1, 0, 1.0f, 0, 0, 0, 0, 0, 0, 0, 1);

  // ---- in-place log-softmax over vocab ----
  logsoftmax_k<<<Tt, blk, 0, stream>>>(out_lp);

  (void)in_sizes; (void)n_in; (void)out_size; (void)ws_size;
}
